// SphericalAttention_12369505813118
// MI455X (gfx1250) — matrix operand order repaired and checked by reading, not compiled, not run on hardware
//
#include <hip/hip_runtime.h>
#include <hip/hip_bf16.h>
#include <cstdint>

// ---------------------------------------------------------------------------
// GQA attention layer for MI455X (gfx1250, wave32, WMMA 16x16x32 bf16)
// B=2 S=4096 D=1024 H=16 KV=4 HD=64, full (non-causal) softmax attention.
// Double-buffered LDS staging via TDM (tensor_load_to_lds) / async-to-LDS.
// ---------------------------------------------------------------------------

typedef __attribute__((ext_vector_type(16))) __bf16 v16bf;
typedef __attribute__((ext_vector_type(8)))  float  v8f;
typedef __attribute__((ext_vector_type(4)))  unsigned int u32x4;
typedef __attribute__((ext_vector_type(8)))  int    i32x8;
typedef __attribute__((ext_vector_type(4)))  int    i32x4;

#define B_   2
#define S_   4096
#define D_   1024
#define H_   16
#define KV_  4
#define HD_  64
#define G_   (H_ / KV_)
#define SCALE_ 0.1f           // HD^-0.5 * 0.8 = 0.8/8

// ---- gfx1250 data-mover feature detection (device pass only) --------------
#if defined(__HIP_DEVICE_COMPILE__) && \
    __has_builtin(__builtin_amdgcn_global_load_async_to_lds_b128) && \
    __has_builtin(__builtin_amdgcn_s_wait_asynccnt)
#define USE_ASYNC 1
#else
#define USE_ASYNC 0
#endif

#if defined(__HIP_DEVICE_COMPILE__) && \
    __has_builtin(__builtin_amdgcn_tensor_load_to_lds) && \
    __has_builtin(__builtin_amdgcn_s_wait_tensorcnt)
#define USE_TDM 1
#else
#define USE_TDM 0
#endif

#if defined(__HIP_DEVICE_COMPILE__)
// Typed address-space pointer casts for the async-to-LDS builtin
// (signature: void(v4i32 AS1*, v4i32 AS3*, imm offset, imm cpol)).
#define GLOBAL_V4(p) ((__attribute__((address_space(1))) i32x4*)(p))
#define LDS_V4(p)    ((__attribute__((address_space(3))) i32x4*)(p))
__device__ __forceinline__ unsigned lds_byte_off(const void* p) {
  return (unsigned)(uintptr_t)(__attribute__((address_space(3))) const void*)(p);
}
#endif

__device__ __forceinline__ v8f wmma_bf16(v16bf a, v16bf b, v8f c) {
  // D = A(16x32 bf16) * B(32x16 bf16) + C(16x16 f32)
  return __builtin_amdgcn_wmma_f32_16x16x32_bf16(false, a, false, b,
                                                 (short)0, c, false, false);
}

__device__ __forceinline__ v8f vzero8() {
  v8f v;
#pragma unroll
  for (int i = 0; i < 8; ++i) v[i] = 0.0f;
  return v;
}

#if USE_TDM
// One TDM descriptor moves a 32-row x 64-elem bf16 tile (row stride 256 elems)
// from global to LDS, inserting 8 DWORDs of padding every 32 DWORDs so the
// LDS image lands in a [32][80] layout (bank-conflict padding done by the DMA).
__device__ __forceinline__ void tdm_load_tile_32x64(const __bf16* gsrc,
                                                    unsigned lds_byte) {
  unsigned long long ga = (unsigned long long)(uintptr_t)gsrc;
  u32x4 g0;
  g0[0] = 1u;                                         // count = 1 (valid D#)
  g0[1] = lds_byte;                                   // lds_addr (bytes)
  g0[2] = (unsigned)(ga & 0xffffffffu);               // global_addr[31:0]
  g0[3] = (unsigned)((ga >> 32) & 0x01ffffffu)        // global_addr[56:32]
        | (2u << 30);                                 // type = 2 ("image")
  i32x8 g1;
  g1[0] = (int)((1u << 16)      // data_size   = 1 -> 2 bytes
              | (1u << 20)      // pad_enable
              | (4u << 22)      // pad_interval: 32 DWORDs
              | (7u << 25));    // pad_amount  : 8 DWORDs
  g1[1] = (int)(64u << 16);     // tensor_dim0 = 64   (bits 63:48)
  g1[2] = (int)(32u << 16);     // tensor_dim1 = 32   (bits 95:80)
  g1[3] = (int)(64u << 16);     // tile_dim0   = 64   (bits 127:112)
  g1[4] = (int)32;              // tile_dim1   = 32   (bits 143:128)
  g1[5] = (int)(KV_ * HD_);     // tensor_dim0_stride = 256 (bits 191:160)
  g1[6] = 0;
  g1[7] = 0;
  i32x4 z4; z4[0] = z4[1] = z4[2] = z4[3] = 0;
  i32x8 z8;
#pragma unroll
  for (int i = 0; i < 8; ++i) z8[i] = 0;
  __builtin_amdgcn_tensor_load_to_lds(g0, g1, z4, z4, z8, 0);
}
#endif

// ---------------------------------------------------------------------------
// fp32 -> bf16 conversion (vectorized by 4)
// ---------------------------------------------------------------------------
__global__ __launch_bounds__(256)
void cvt_f32_bf16_kernel(const float* __restrict__ src, __bf16* __restrict__ dst,
                         long n) {
  long i = ((long)blockIdx.x * 256 + threadIdx.x) * 4;
  if (i + 3 < n) {
    float4 v = *(const float4*)(src + i);
    dst[i + 0] = (__bf16)v.x;
    dst[i + 1] = (__bf16)v.y;
    dst[i + 2] = (__bf16)v.z;
    dst[i + 3] = (__bf16)v.w;
  }
}

// ---------------------------------------------------------------------------
// Tiled bf16 GEMM: C[M,N] = A[M,K] @ B[K,N] (row-major bf16 in, f32/bf16 out)
// Workgroup tile 128x128, 8 waves (4x2); wave tile 32x64 = 2x4 WMMA tiles.
// Double-buffered LDS tiles, staged with async-to-LDS when available.
// ---------------------------------------------------------------------------
template <bool OUT_F32>
__global__ __launch_bounds__(256)
void gemm_bf16_kernel(const __bf16* __restrict__ A, const __bf16* __restrict__ Bm,
                      float* __restrict__ Cf, __bf16* __restrict__ Cb,
                      int M, int N, int K) {
  __shared__ __bf16 As[2][128][48];   // 96B row stride
  __shared__ __bf16 Bs[2][32][144];   // 288B row stride

  const int tid  = threadIdx.x;
  const int lane = tid & 31;
  const int wv   = tid >> 5;
  const int hf   = lane >> 4;
  const int l16  = lane & 15;
  const int wm   = wv >> 1;
  const int wn   = wv & 1;
  const int m0   = blockIdx.y * 128;
  const int n0   = blockIdx.x * 128;

  v8f acc[2][4];
#pragma unroll
  for (int ms = 0; ms < 2; ++ms)
#pragma unroll
    for (int ns = 0; ns < 4; ++ns) acc[ms][ns] = vzero8();

  auto stage = [&](int kk, int nb) {
#pragma unroll
    for (int it = 0; it < 2; ++it) {
      int q  = tid + 256 * it;
      int ra = q >> 2, ca = (q & 3) * 8;
      int rb = q >> 4, cb = (q & 15) * 8;
      const __bf16* ga = &A[(long)(m0 + ra) * K + kk + ca];
      const __bf16* gb = &Bm[(long)(kk + rb) * N + n0 + cb];
#if USE_ASYNC
      __builtin_amdgcn_global_load_async_to_lds_b128(GLOBAL_V4(ga),
                                                     LDS_V4(&As[nb][ra][ca]), 0, 0);
      __builtin_amdgcn_global_load_async_to_lds_b128(GLOBAL_V4(gb),
                                                     LDS_V4(&Bs[nb][rb][cb]), 0, 0);
#else
      *(uint4*)&As[nb][ra][ca] = *(const uint4*)ga;
      *(uint4*)&Bs[nb][rb][cb] = *(const uint4*)gb;
#endif
    }
#if !USE_ASYNC
    if (kk + 32 < K) {
      __builtin_prefetch(&A[(long)(m0 + (tid >> 2)) * K + kk + 32], 0, 0);
      __builtin_prefetch(&Bm[(long)(kk + 32 + (tid >> 4)) * N + n0], 0, 0);
    }
#endif
  };
  auto stage_wait = [&]() {
#if USE_ASYNC
    __builtin_amdgcn_s_wait_asynccnt(0);
#endif
  };

  stage(0, 0);
  stage_wait();
  __syncthreads();
  int cur = 0;

  for (int kk = 0; kk < K; kk += 32) {
    if (kk + 32 < K) stage(kk + 32, cur ^ 1);   // overlap next tile with math

    // A fragment: lane row=l16; element e -> k = (e&7) + 16*(e>>3) + 8*hf
    v16bf af[2];
#pragma unroll
    for (int ms = 0; ms < 2; ++ms) {
      int row = wm * 32 + ms * 16 + l16;
#pragma unroll
      for (int e = 0; e < 16; ++e)
        af[ms][e] = As[cur][row][(e & 7) + 16 * (e >> 3) + 8 * hf];
    }
    // B fragment: lane col=l16; element e -> k = (e&7) + 16*(e>>3) + 8*hf
    v16bf bfr[4];
#pragma unroll
    for (int ns = 0; ns < 4; ++ns) {
      int col = wn * 64 + ns * 16 + l16;
#pragma unroll
      for (int e = 0; e < 16; ++e)
        bfr[ns][e] = Bs[cur][(e & 7) + 16 * (e >> 3) + 8 * hf][col];
    }
#pragma unroll
    for (int ms = 0; ms < 2; ++ms)
#pragma unroll
      for (int ns = 0; ns < 4; ++ns)
        acc[ms][ns] = wmma_bf16(af[ms], bfr[ns], acc[ms][ns]);

    stage_wait();
    __syncthreads();
    cur ^= 1;
  }

  // C layout: element i -> row = i + 8*hf, col = l16
#pragma unroll
  for (int ms = 0; ms < 2; ++ms)
#pragma unroll
    for (int ns = 0; ns < 4; ++ns)
#pragma unroll
      for (int i = 0; i < 8; ++i) {
        long row = m0 + wm * 32 + ms * 16 + i + 8 * hf;
        long col = n0 + wn * 64 + ns * 16 + l16;
        if constexpr (OUT_F32) Cf[row * N + col] = acc[ms][ns][i];
        else                   Cb[row * N + col] = (__bf16)acc[ms][ns][i];
      }
}

// ---------------------------------------------------------------------------
// RoPE in place on bf16 Q or K laid out as (B*S, nh*HD) row-major.
// ---------------------------------------------------------------------------
__global__ __launch_bounds__(256)
void rope_kernel(__bf16* __restrict__ x, int nh) {
  long i = (long)blockIdx.x * 256 + threadIdx.x;   // over B*S*nh*32
  int  d = (int)(i & 31);
  long r = i >> 5;
  int  h = (int)(r % nh);
  long row = r / nh;                               // b*S + s
  int  s = (int)(row & (S_ - 1));
  float inv = exp2f(-(float)d * (13.287712379549449f / 32.0f));
  float ang = (float)s * inv;
  float sn, cs;
  __sincosf(ang, &sn, &cs);
  __bf16* p = x + row * (long)(nh * HD_) + (long)h * HD_;
  float x1 = (float)p[d];
  float x2 = (float)p[d + 32];
  p[d]      = (__bf16)(x1 * cs - x2 * sn);
  p[d + 32] = (__bf16)(x2 * cs + x1 * sn);
}

// ---------------------------------------------------------------------------
// Flash attention (full, non-causal). Grid: (S/64, H, B), 128 threads = 4 waves.
// Wave w owns query rows [qb*64 + 16w, +16). KV tiles of 32 double-buffered in
// LDS, staged by the Tensor Data Mover (wave0: K, wave1: V) when available.
// ---------------------------------------------------------------------------
__global__ __launch_bounds__(128)
void attn_kernel(const __bf16* __restrict__ Q, const __bf16* __restrict__ Kc,
                 const __bf16* __restrict__ Vc, __bf16* __restrict__ O) {
  __shared__ __bf16 Kt[2][32][80];    // 32 keys x 64 dims (+8-DWORD row pad)
  __shared__ __bf16 Vt[2][32][80];
  __shared__ __bf16 Pt[4][16][40];    // per-wave probs 16 rows x 32 keys

  const int tid  = threadIdx.x;
  const int lane = tid & 31;
  const int w    = tid >> 5;
  const int hf   = lane >> 4;
  const int l16  = lane & 15;
  const int h    = blockIdx.y;
  const int b    = blockIdx.z;
  const int kvh  = h / G_;
  const long qrow0 = (long)b * S_ + (long)blockIdx.x * 64 + w * 16;

  // Q fragments held in registers across the whole kv loop
  v16bf qf[2];
  {
    const __bf16* qp = Q + (qrow0 + l16) * (long)(H_ * HD_) + h * HD_;
#pragma unroll
    for (int c = 0; c < 2; ++c)
#pragma unroll
      for (int e = 0; e < 16; ++e)
        qf[c][e] = qp[c * 32 + (e & 7) + 16 * (e >> 3) + 8 * hf];
  }

  v8f o0[4];
#pragma unroll
  for (int f = 0; f < 4; ++f) o0[f] = vzero8();
  float m[8], l[8];
#pragma unroll
  for (int i = 0; i < 8; ++i) { m[i] = -1e30f; l[i] = 0.0f; }

  const __bf16* kb = Kc + (long)b * S_ * (KV_ * HD_) + kvh * HD_;
  const __bf16* vb = Vc + (long)b * S_ * (KV_ * HD_) + kvh * HD_;
  const int ldr = tid >> 3;
  const int ldc = (tid & 7) * 8;

  auto stage = [&](int kt, int nb) {
#if USE_TDM
    if (w == 0)
      tdm_load_tile_32x64(&kb[(long)kt * (KV_ * HD_)],
                          lds_byte_off(&Kt[nb][0][0]));
    else if (w == 1)
      tdm_load_tile_32x64(&vb[(long)kt * (KV_ * HD_)],
                          lds_byte_off(&Vt[nb][0][0]));
#else
#pragma unroll
    for (int rr = 0; rr < 2; ++rr) {
      int row = ldr + rr * 16;
      const __bf16* gk = &kb[(long)(kt + row) * (KV_ * HD_) + ldc];
      const __bf16* gv = &vb[(long)(kt + row) * (KV_ * HD_) + ldc];
#if USE_ASYNC
      __builtin_amdgcn_global_load_async_to_lds_b128(GLOBAL_V4(gk),
                                                     LDS_V4(&Kt[nb][row][ldc]), 0, 0);
      __builtin_amdgcn_global_load_async_to_lds_b128(GLOBAL_V4(gv),
                                                     LDS_V4(&Vt[nb][row][ldc]), 0, 0);
#else
      *(uint4*)&Kt[nb][row][ldc] = *(const uint4*)gk;
      *(uint4*)&Vt[nb][row][ldc] = *(const uint4*)gv;
#endif
    }
#endif
  };
  auto stage_wait = [&]() {
#if USE_TDM
    __builtin_amdgcn_s_wait_tensorcnt(0);
#elif USE_ASYNC
    __builtin_amdgcn_s_wait_asynccnt(0);
#endif
  };

  stage(0, 0);
  stage_wait();
  __syncthreads();
  int cur = 0;

  for (int kt = 0; kt < S_; kt += 32) {
    if (kt + 32 < S_) stage(kt + 32, cur ^ 1);   // DMA next tile during math

    // ---- scores: 16 q-rows x 32 keys ----
    v8f s0 = vzero8(), s1 = vzero8();
#pragma unroll
    for (int c = 0; c < 2; ++c) {
      v16bf b0, b1;
#pragma unroll
      for (int e = 0; e < 16; ++e) {
        int dim = c * 32 + (e & 7) + 16 * (e >> 3) + 8 * hf;
        b0[e] = Kt[cur][l16][dim];        // keys 0..15
        b1[e] = Kt[cur][16 + l16][dim];   // keys 16..31
      }
      s0 = wmma_bf16(qf[c], b0, s0);
      s1 = wmma_bf16(qf[c], b1, s1);
    }

    // ---- online softmax (row lives across a 16-lane half-wave group) ----
#pragma unroll
    for (int i = 0; i < 8; ++i) {
      float a  = s0[i] * SCALE_;
      float bb = s1[i] * SCALE_;
      float mx = fmaxf(a, bb);
#pragma unroll
      for (int off = 1; off < 16; off <<= 1)
        mx = fmaxf(mx, __shfl_xor(mx, off, 32));
      float mn = fmaxf(m[i], mx);
      float p0 = __expf(a - mn);
      float p1 = __expf(bb - mn);
      float sm = p0 + p1;
#pragma unroll
      for (int off = 1; off < 16; off <<= 1)
        sm += __shfl_xor(sm, off, 32);
      float al = __expf(m[i] - mn);
      m[i] = mn;
      l[i] = l[i] * al + sm;
#pragma unroll
      for (int f = 0; f < 4; ++f) o0[f][i] *= al;
      int r = i + 8 * hf;
      Pt[w][r][l16]      = (__bf16)p0;
      Pt[w][r][16 + l16] = (__bf16)p1;
    }
    __syncthreads();

    // ---- PV: O(16x64) += P(16x32) @ V(32x64) ----
#pragma unroll
    for (int c = 0; c < 4; ++c) {
      v16bf pf, vf;
#pragma unroll
      for (int e = 0; e < 16; ++e) {
        pf[e] = Pt[w][l16][(e & 7) + 16 * (e >> 3) + 8 * hf];  // A frag
        vf[e] = Vt[cur][(e & 7) + 16 * (e >> 3) + 8 * hf][c * 16 + l16];            // B frag
      }
      o0[c] = wmma_bf16(pf, vf, o0[c]);
    }

    stage_wait();
    __syncthreads();
    cur ^= 1;
  }

  // Normalize and store (bf16) attention output (B*S, H*HD)
#pragma unroll
  for (int f = 0; f < 4; ++f)
#pragma unroll
    for (int i = 0; i < 8; ++i) {
      long row = qrow0 + i + 8 * hf;
      O[row * (long)(H_ * HD_) + h * HD_ + f * 16 + l16] =
          (__bf16)(o0[f][i] / l[i]);
    }
}

// ---------------------------------------------------------------------------
// Host-side orchestration
// ---------------------------------------------------------------------------
extern "C" void kernel_launch(void* const* d_in, const int* in_sizes, int n_in,
                              void* d_out, int out_size, void* d_ws, size_t ws_size,
                              hipStream_t stream) {
  (void)in_sizes; (void)n_in; (void)out_size; (void)ws_size;
  const float* hs = (const float*)d_in[0];
  const float* wq = (const float*)d_in[1];
  const float* wk = (const float*)d_in[2];
  const float* wv = (const float*)d_in[3];
  const float* wo = (const float*)d_in[4];

  char* p = (char*)d_ws;
  auto alloc = [&](size_t elems) {
    __bf16* r = (__bf16*)p;
    p += ((elems * 2 + 255) / 256) * 256;
    return r;
  };
  __bf16* hbf = alloc((size_t)B_ * S_ * D_);        // 16 MB
  __bf16* wqb = alloc((size_t)D_ * H_ * HD_);       //  2 MB
  __bf16* wkb = alloc((size_t)D_ * KV_ * HD_);      // .5 MB
  __bf16* wvb = alloc((size_t)D_ * KV_ * HD_);      // .5 MB
  __bf16* wob = alloc((size_t)H_ * HD_ * D_);       //  2 MB
  __bf16* Qb  = alloc((size_t)B_ * S_ * H_ * HD_);  // 16 MB
  __bf16* Kb  = alloc((size_t)B_ * S_ * KV_ * HD_); //  4 MB
  __bf16* Vb  = alloc((size_t)B_ * S_ * KV_ * HD_); //  4 MB
  __bf16* Ob  = alloc((size_t)B_ * S_ * H_ * HD_);  // 16 MB  (~62 MB total)

  auto cvt = [&](const float* s, __bf16* d, long n) {
    cvt_f32_bf16_kernel<<<dim3((unsigned)(n / 1024)), 256, 0, stream>>>(s, d, n);
  };
  cvt(hs, hbf, (long)B_ * S_ * D_);
  cvt(wq, wqb, (long)D_ * H_ * HD_);
  cvt(wk, wkb, (long)D_ * KV_ * HD_);
  cvt(wv, wvb, (long)D_ * KV_ * HD_);
  cvt(wo, wob, (long)H_ * HD_ * D_);

  const int M = B_ * S_;  // 8192
  gemm_bf16_kernel<false><<<dim3((H_ * HD_) / 128, M / 128), 256, 0, stream>>>(
      hbf, wqb, nullptr, Qb, M, H_ * HD_, D_);
  gemm_bf16_kernel<false><<<dim3((KV_ * HD_) / 128, M / 128), 256, 0, stream>>>(
      hbf, wkb, nullptr, Kb, M, KV_ * HD_, D_);
  gemm_bf16_kernel<false><<<dim3((KV_ * HD_) / 128, M / 128), 256, 0, stream>>>(
      hbf, wvb, nullptr, Vb, M, KV_ * HD_, D_);

  rope_kernel<<<dim3((unsigned)(((long)M * H_ * 32) / 256)), 256, 0, stream>>>(Qb, H_);
  rope_kernel<<<dim3((unsigned)(((long)M * KV_ * 32) / 256)), 256, 0, stream>>>(Kb, KV_);

  attn_kernel<<<dim3(S_ / 64, H_, B_), 128, 0, stream>>>(Qb, Kb, Vb, Ob);

  gemm_bf16_kernel<true><<<dim3(D_ / 128, M / 128), 256, 0, stream>>>(
      Ob, wob, (float*)d_out, nullptr, M, D_, H_ * HD_);
}
